// PointSetEmbedding_64707977281967
// MI455X (gfx1250) — matrix operand order repaired and checked by reading, not compiled, not run on hardware
//
#include <hip/hip_runtime.h>
#include <hip/hip_bf16.h>
#include <math.h>

typedef __attribute__((ext_vector_type(16))) __bf16 v16bf;
typedef __attribute__((ext_vector_type(8)))  __bf16 v8bf;
typedef __attribute__((ext_vector_type(8)))  float  v8f;
typedef __attribute__((ext_vector_type(4)))  int    v4i;

#define B_    4
#define N_    16384
#define S_    1024      // n_point
#define K_    64        // n_sample
#define CIN   259       // 256 + 3 (channel-permuted: points first, xyz last)
#define CPAD  288       // K padded to 9 * 32 for bf16 WMMA
#define ASTR  296       // LDS row stride (bf16 elems) for A tile (592B = 37*16)
#define HSTR  528       // LDS row stride (bf16 elems) for H tile (1056B = 66*16)
#define O1    512
#define RAD2  0.04f     // radius^2

#if __has_builtin(__builtin_amdgcn_global_load_async_to_lds_b128)
#define HAVE_ASYNC_LDS 1
#else
#define HAVE_ASYNC_LDS 0
#endif

static __device__ __forceinline__ unsigned short f2bf(float x) {
    unsigned u = __float_as_uint(x);
    unsigned r = u + 0x7fffu + ((u >> 16) & 1u);   // round-to-nearest-even
    return (unsigned short)(r >> 16);
}

static __device__ __forceinline__ float swish(float x) {
    return x / (1.0f + __expf(-x));
}

// ---------------------------------------------------------------------------
// Kernel 1a: convert weights to bf16.
// Channel permutation: feat channel c<256 -> original 3+c (points),
//                      c in [256,259) -> original c-256 (normed xyz), rest 0.
// ---------------------------------------------------------------------------
__global__ __launch_bounds__(256) void convert_weights_kernel(
    const float* __restrict__ w1, const float* __restrict__ w2,
    unsigned short* __restrict__ w1p, unsigned short* __restrict__ w2b) {
    int t = blockIdx.x * blockDim.x + threadIdx.x;
    const int n1 = O1 * CPAD;                 // 147456
    if (t < n1) {
        int o = t / CPAD, c = t % CPAD;
        float v = 0.0f;
        if (c < 256)      v = w1[o * CIN + 3 + c];
        else if (c < CIN) v = w1[o * CIN + (c - 256)];
        w1p[t] = (c < CIN) ? f2bf(v) : (unsigned short)0;
    } else {
        int u = t - n1;
        if (u < O1 * O1) w2b[u] = f2bf(w2[u]);
    }
}

// ---------------------------------------------------------------------------
// Kernel 1b: transpose+convert points (B,256,N) f32 -> (B,N,256) bf16.
// Makes every gathered token a contiguous 512B row (16B-aligned copies).
// ---------------------------------------------------------------------------
__global__ __launch_bounds__(256) void convert_points_kernel(
    const float* __restrict__ points, unsigned short* __restrict__ ptsT) {
    size_t id = (size_t)blockIdx.x * 256 + threadIdx.x;   // b*N*256 total
    int c = (int)(id & 255);
    size_t bn = id >> 8;            // b*N + n
    int n = (int)(bn & (N_ - 1));
    int b = (int)(bn >> 14);
    ptsT[id] = f2bf(points[((size_t)b * 256 + c) * N_ + n]);
}

// ---------------------------------------------------------------------------
// Kernel 2: farthest point sampling. One 1024-thread workgroup per batch.
// xyz coords + running min-distance kept entirely in LDS (256KB < 320KB/WGP).
// ---------------------------------------------------------------------------
__global__ __launch_bounds__(1024) void fps_kernel(
    const float* __restrict__ xyz, int* __restrict__ fps_idx) {
    extern __shared__ char smem[];
    float* xs   = (float*)smem;
    float* ys   = xs + N_;
    float* zs   = ys + N_;
    float* dist = zs + N_;
    float* rval = dist + N_;               // 32 per-wave argmax values
    int*   ridx = (int*)(rval + 32);       // 32 per-wave argmax indices
    int*   cur  = (int*)(ridx + 32);       // current farthest index

    const int b = blockIdx.x;
    const int t = threadIdx.x;
    const int lane = t & 31, wave = t >> 5;
    const float* x0 = xyz + (size_t)b * 3 * N_;
    const float* x1 = x0 + N_;
    const float* x2 = x1 + N_;

    for (int n = t; n < N_; n += 1024) {
        xs[n] = x0[n]; ys[n] = x1[n]; zs[n] = x2[n];
        dist[n] = 1e10f;
    }
    if (t == 0) *cur = b;                  // arange(B) % N
    __syncthreads();

    for (int step = 0; step < S_; ++step) {
        int f = *cur;
        if (t == 0) fps_idx[b * S_ + step] = f;
        float cx = xs[f], cy = ys[f], cz = zs[f];

        float bv = -1.0f; int bi = 0;
        for (int n = t; n < N_; n += 1024) {
            float dx = xs[n] - cx, dy = ys[n] - cy, dz = zs[n] - cz;
            float d = dx * dx + dy * dy + dz * dz;
            d = fminf(dist[n], d);
            dist[n] = d;
            if (d > bv) { bv = d; bi = n; }      // lowest n wins on ties
        }
        // wave32 argmax reduction (first-occurrence tie-break)
        for (int off = 16; off > 0; off >>= 1) {
            float ov = __shfl_down(bv, off, 32);
            int   oi = __shfl_down(bi, off, 32);
            if (ov > bv || (ov == bv && oi < bi)) { bv = ov; bi = oi; }
        }
        if (lane == 0) { rval[wave] = bv; ridx[wave] = bi; }
        __syncthreads();
        if (wave == 0) {
            float v = rval[lane]; int i = ridx[lane];
            for (int off = 16; off > 0; off >>= 1) {
                float ov = __shfl_down(v, off, 32);
                int   oi = __shfl_down(i, off, 32);
                if (ov > v || (ov == v && oi < i)) { v = ov; i = oi; }
            }
            if (lane == 0) *cur = i;
        }
        __syncthreads();
    }
}

// ---------------------------------------------------------------------------
// Kernel 3: ball query. One wave per centroid; ballot/popcount prefix
// collects the first K_ in-radius point indices in ascending order.
// ---------------------------------------------------------------------------
__global__ __launch_bounds__(256) void ballquery_kernel(
    const float* __restrict__ xyz, const int* __restrict__ fps_idx,
    int* __restrict__ ball_idx) {
    const int wave = threadIdx.x >> 5, lane = threadIdx.x & 31;
    const int g = blockIdx.x * 8 + wave;     // 0..4095
    const int b = g >> 10, s = g & 1023;
    const float* x0 = xyz + (size_t)b * 3 * N_;
    const float* x1 = x0 + N_;
    const float* x2 = x1 + N_;

    const int nc = fps_idx[b * S_ + s];
    const float cx = x0[nc], cy = x1[nc], cz = x2[nc];
    int* out = ball_idx + (size_t)g * K_;

    int cnt = 0, first = -1;
    for (int base = 0; base < N_ && cnt < K_; base += 32) {
        int n = base + lane;
        float dx = x0[n] - cx, dy = x1[n] - cy, dz = x2[n] - cz;
        bool pred = (dx * dx + dy * dy + dz * dz) <= RAD2;
        unsigned mask = (unsigned)__ballot(pred);
        if (first < 0 && mask) first = base + (__ffs(mask) - 1);
        int before = __popc(mask & ((1u << lane) - 1u));
        int slot = cnt + before;
        if (pred && slot < K_) out[slot] = n;
        cnt += __popc(mask);
    }
    if (cnt > K_) cnt = K_;
    for (int j = cnt + lane; j < K_; j += 32) out[j] = first;  // pad with first
}

// ---------------------------------------------------------------------------
// Helper: build a v16bf A-fragment from two 16B LDS chunks
// ---------------------------------------------------------------------------
static __device__ __forceinline__ v16bf load_afrag(const unsigned short* ap) {
    v8bf alo = *(const v8bf*)ap;
    v8bf ahi = *(const v8bf*)(ap + 8);
    v16bf av;
    #pragma unroll
    for (int j = 0; j < 8; ++j) { av[j] = alo[j]; av[j + 8] = ahi[j]; }
    return av;
}

// ---------------------------------------------------------------------------
// Kernel 4: fused gather + MLP1 + MLP2 + mean. One workgroup (8 waves) per
// (b,s) group. A-tile (64 x 288 bf16) and H-tile (64 x 512 bf16) live in LDS.
// Weight fragments are loaded once per ks and reused across 4 M-tiles.
// ---------------------------------------------------------------------------
__global__ __launch_bounds__(256) void group_mlp_kernel(
    const float* __restrict__ xyz, const float* __restrict__ points,
    const unsigned short* __restrict__ ptsT,   // may be null -> f32 fallback
    const unsigned short* __restrict__ w1p, const float* __restrict__ b1,
    const unsigned short* __restrict__ w2b, const float* __restrict__ b2,
    const int* __restrict__ fps_idx, const int* __restrict__ ball_idx,
    float* __restrict__ out) {
    extern __shared__ char smem[];
    unsigned short* A = (unsigned short*)smem;        // K_ x ASTR bf16
    unsigned short* H = A + K_ * ASTR;                // K_ x HSTR bf16
    int* gi = (int*)(H + K_ * HSTR);                  // 64 gathered indices

    const int t = threadIdx.x;
    const int g = blockIdx.x;
    const int b = g >> 10, s = g & 1023;
    const float* x0  = xyz + (size_t)b * 3 * N_;
    const float* x1  = x0 + N_;
    const float* x2  = x1 + N_;

    if (t < K_) gi[t] = ball_idx[(size_t)g * K_ + t];
    __syncthreads();

    // zero-pad channels [259, 288)
    for (int i = t; i < K_ * 32; i += 256) {
        int k = i >> 5, c = 256 + (i & 31);
        if (c >= CIN) A[k * ASTR + c] = 0;
    }
    // gather normed xyz -> channels 256..258
    if (t < K_) {
        int n = gi[t];
        int ncen = fps_idx[b * S_ + s];
        A[t * ASTR + 256] = f2bf(x0[n] - x0[ncen]);
        A[t * ASTR + 257] = f2bf(x1[n] - x1[ncen]);
        A[t * ASTR + 258] = f2bf(x2[n] - x2[ncen]);
    }
    // gather point features -> channels 0..255
    if (ptsT) {
        // bf16 token-major: contiguous 512B rows, 16B-aligned copies
        for (int i = t; i < K_ * 32; i += 256) {
            int k = i >> 5, seg = i & 31;
            const unsigned short* src =
                ptsT + ((size_t)b * N_ + gi[k]) * 256 + seg * 8;
            unsigned short* dst = &A[k * ASTR + seg * 8];
#if HAVE_ASYNC_LDS
            __builtin_amdgcn_global_load_async_to_lds_b128(
                (__attribute__((address_space(1))) v4i*)src,
                (__attribute__((address_space(3))) v4i*)dst, 0, 0);
#else
            *(uint4*)dst = *(const uint4*)src;
#endif
        }
#if HAVE_ASYNC_LDS
#if __has_builtin(__builtin_amdgcn_s_wait_asynccnt)
        __builtin_amdgcn_s_wait_asynccnt(0);
#else
        asm volatile("s_wait_asynccnt 0x0" ::: "memory");
#endif
#endif
    } else {
        const float* pts = points + (size_t)b * 256 * N_;
        for (int i = t; i < K_ * 256; i += 256) {
            int c = i >> 6, k = i & 63;
            A[k * ASTR + c] = f2bf(pts[c * N_ + gi[k]]);
        }
    }
    __syncthreads();

    const int lane = t & 31, wave = t >> 5;
    const int hi = lane >> 4, l16 = lane & 15;

    // ---- GEMM1: H[64,512] = swish(A[64,288] * W1p^T + b1), bf16 into LDS --
    for (int nq = 0; nq < 4; ++nq) {
        const int nt = wave * 4 + nq;
        const int o  = nt * 16 + l16;
        v8f acc[4] = {};
        for (int ks = 0; ks < CPAD / 32; ++ks) {
            v16bf bvv = *(const v16bf*)(w1p + o * CPAD + ks * 32 + hi * 16);
            #pragma unroll
            for (int mt = 0; mt < 4; ++mt) {
                v16bf av = load_afrag(A + (mt * 16 + l16) * ASTR + ks * 32 + hi * 16);
                acc[mt] = __builtin_amdgcn_wmma_f32_16x16x32_bf16(
                              false, av, false, bvv, (short)0, acc[mt], false, false);
            }
        }
        float bias = b1[o];
        #pragma unroll
        for (int mt = 0; mt < 4; ++mt) {
            #pragma unroll
            for (int r = 0; r < 8; ++r) {
                float y = swish(acc[mt][r] + bias);
                H[(mt * 16 + r + 8 * hi) * HSTR + o] = f2bf(y);
            }
        }
    }
    __syncthreads();

    // ---- GEMM2 + mean over the 64 samples ----
    for (int nq = 0; nq < 4; ++nq) {
        const int nt = wave * 4 + nq;
        const int o  = nt * 16 + l16;
        const float bias = b2[o];
        v8f acc[4] = {};
        for (int ks = 0; ks < O1 / 32; ++ks) {
            v16bf bvv = *(const v16bf*)(w2b + o * O1 + ks * 32 + hi * 16);
            #pragma unroll
            for (int mt = 0; mt < 4; ++mt) {
                v16bf av = load_afrag(H + (mt * 16 + l16) * HSTR + ks * 32 + hi * 16);
                acc[mt] = __builtin_amdgcn_wmma_f32_16x16x32_bf16(
                              false, av, false, bvv, (short)0, acc[mt], false, false);
            }
        }
        float colp = 0.0f;
        #pragma unroll
        for (int mt = 0; mt < 4; ++mt)
            #pragma unroll
            for (int r = 0; r < 8; ++r)
                colp += swish(acc[mt][r] + bias);
        colp += __shfl_xor(colp, 16, 32);   // fold the two 16-lane halves
        if (hi == 0)
            out[((size_t)b * 512 + o) * 1024 + s] = colp * (1.0f / 64.0f);
    }
}

// ---------------------------------------------------------------------------
extern "C" void kernel_launch(void* const* d_in, const int* in_sizes, int n_in,
                              void* d_out, int out_size, void* d_ws, size_t ws_size,
                              hipStream_t stream) {
    const float* xyz    = (const float*)d_in[0];
    const float* points = (const float*)d_in[1];
    const float* w1     = (const float*)d_in[2];
    const float* b1     = (const float*)d_in[3];
    const float* w2     = (const float*)d_in[4];
    const float* b2     = (const float*)d_in[5];
    float* out = (float*)d_out;

    char* ws = (char*)d_ws;
    const size_t OFF_BALL = 16384;                       // fps: 16KB
    const size_t OFF_W1P  = OFF_BALL + 1048576;          // ball: 1MB
    const size_t OFF_W2B  = OFF_W1P + (size_t)O1 * CPAD * 2;   // w1p: 288KB
    const size_t OFF_PTST = OFF_W2B + (size_t)O1 * O1 * 2;     // w2b: 512KB
    const size_t NEED     = OFF_PTST + (size_t)B_ * N_ * 256 * 2;  // +32MB

    int* fps_idx  = (int*)ws;
    int* ball_idx = (int*)(ws + OFF_BALL);
    unsigned short* w1p = (unsigned short*)(ws + OFF_W1P);
    unsigned short* w2b = (unsigned short*)(ws + OFF_W2B);
    unsigned short* ptsT = (ws_size >= NEED) ? (unsigned short*)(ws + OFF_PTST)
                                             : (unsigned short*)nullptr;

    // 1a) weights -> bf16 (padded + channel-permuted)
    int ntot = O1 * CPAD + O1 * O1;  // 409600
    convert_weights_kernel<<<(ntot + 255) / 256, 256, 0, stream>>>(w1, w2, w1p, w2b);

    // 1b) points -> token-major bf16 (if workspace allows)
    if (ptsT) {
        int nblk = (B_ * N_ * 256) / 256;   // 65536
        convert_points_kernel<<<nblk, 256, 0, stream>>>(points, ptsT);
    }

    // 2) FPS: coords + distances in LDS (4*N_ floats + reduction scratch)
    size_t fps_smem = (size_t)4 * N_ * 4 + 32 * 8 + 16;
    fps_kernel<<<B_, 1024, fps_smem, stream>>>(xyz, fps_idx);

    // 3) ball query: one wave per centroid
    ballquery_kernel<<<(B_ * S_) / 8, 256, 0, stream>>>(xyz, fps_idx, ball_idx);

    // 4) fused gather + two WMMA GEMMs + mean
    size_t g_smem = (size_t)(K_ * ASTR + K_ * HSTR) * 2 + 64 * 4;  // ~105.7 KB
    group_mlp_kernel<<<B_ * S_, 256, g_smem, stream>>>(
        xyz, points, ptsT, w1p, b1, w2b, b2, fps_idx, ball_idx, out);
}
